// GRACE_42958262894593
// MI455X (gfx1250) — matrix operand order repaired and checked by reading, not compiled, not run on hardware
//
#include <hip/hip_runtime.h>
#include <hip/hip_bf16.h>
#include <math.h>

typedef __attribute__((ext_vector_type(16))) __bf16 v16bf;
typedef __attribute__((ext_vector_type(8)))  __bf16 bf16x8;
typedef __attribute__((ext_vector_type(8)))  float  v8f;

#define INV_TAU 2.0f
#define NSPLIT 8

// ---------- WMMA helpers ----------
__device__ __forceinline__ v8f wmma_bf16(v16bf a, v16bf b, v8f c) {
    return __builtin_amdgcn_wmma_f32_16x16x32_bf16(false, a, false, b, (short)0, c, false, false);
}

// A-fragment (16x32 bf16): lane L holds row m=L&15.
// K indices: elems 0..7  -> k0 + hi*8 + e        (contiguous 8)
//            elems 8..15 -> k0 + 16 + hi*8 + e-8 (contiguous 8)
__device__ __forceinline__ v16bf load_a_frag(const __bf16* rowptr, int k0, int hi) {
    union { v16bf v; bf16x8 h[2]; } u;
    u.h[0] = *(const bf16x8*)(rowptr + k0 + hi * 8);
    u.h[1] = *(const bf16x8*)(rowptr + k0 + 16 + hi * 8);
    return u.v;
}

// B-fragment (32x16 bf16): lane L holds col n=L&15 (a row of the [n][k] operand).
// K indices: elems 0..7 -> k0 + hi*8 + e, elems 8..15 -> k0 + 16 + hi*8 + e-8
__device__ __forceinline__ v16bf load_b_frag(const __bf16* rowptr, int k0, int hi) {
    union { v16bf v; bf16x8 h[2]; } u;
    u.h[0] = *(const bf16x8*)(rowptr + k0 + hi * 8);
    u.h[1] = *(const bf16x8*)(rowptr + k0 + 16 + hi * 8);
    return u.v;
}

// ---------- elementwise conversion ----------
__global__ void k_cvt_bf16(const float* __restrict__ src, __bf16* __restrict__ dst, int n) {
    int i = blockIdx.x * blockDim.x + threadIdx.x;
    if (i < n) dst[i] = (__bf16)src[i];
}

// Wc[n*256+k] = W[k*256+n]  (store weights as [outCol][k] so B-frags are contiguous)
__global__ void k_transpose_cvt(const float* __restrict__ W, __bf16* __restrict__ Wc) {
    int n = blockIdx.x;
    int k = threadIdx.x;
    Wc[n * 256 + k] = (__bf16)W[k * 256 + n];
}

// ---------- projection GEMM 1: mid = elu(z @ W1 + b1), bf16 out ----------
__global__ __launch_bounds__(256) void k_proj1(const __bf16* __restrict__ A,
                                               const __bf16* __restrict__ Bc,
                                               const float* __restrict__ bias,
                                               __bf16* __restrict__ mid) {
    int tid = threadIdx.x;
    int wave = tid >> 5, lane = tid & 31;
    int mw = wave & 3, nw = wave >> 2;
    int hi = lane >> 4, ln = lane & 15;
    int gm0 = blockIdx.x * 64 + mw * 16;

    const __bf16* arow = A + (size_t)(gm0 + ln) * 256;
    v8f acc[8] = {};
#pragma unroll
    for (int kc = 0; kc < 8; ++kc) {
        int k0 = kc * 32;
        v16bf af = load_a_frag(arow, k0, hi);
#pragma unroll
        for (int t = 0; t < 8; ++t) {
            const __bf16* brow = Bc + (size_t)(nw * 128 + t * 16 + ln) * 256;
            acc[t] = wmma_bf16(af, load_b_frag(brow, k0, hi), acc[t]);
        }
    }
#pragma unroll
    for (int t = 0; t < 8; ++t) {
        int gc = nw * 128 + t * 16 + ln;
        float bv = bias[gc];
#pragma unroll
        for (int r = 0; r < 8; ++r) {
            int gm = gm0 + hi * 8 + r;
            float x = acc[t][r] + bv;
            x = (x > 0.0f) ? x : (__expf(x) - 1.0f);  // elu
            mid[(size_t)gm * 256 + gc] = (__bf16)x;
        }
    }
}

// ---------- projection GEMM 2: h = mid @ W2 + b2, f32 out ----------
__global__ __launch_bounds__(256) void k_proj2(const __bf16* __restrict__ A,
                                               const __bf16* __restrict__ Bc,
                                               const float* __restrict__ bias,
                                               float* __restrict__ h) {
    int tid = threadIdx.x;
    int wave = tid >> 5, lane = tid & 31;
    int mw = wave & 3, nw = wave >> 2;
    int hi = lane >> 4, ln = lane & 15;
    int gm0 = blockIdx.x * 64 + mw * 16;

    const __bf16* arow = A + (size_t)(gm0 + ln) * 256;
    v8f acc[8] = {};
#pragma unroll
    for (int kc = 0; kc < 8; ++kc) {
        int k0 = kc * 32;
        v16bf af = load_a_frag(arow, k0, hi);
#pragma unroll
        for (int t = 0; t < 8; ++t) {
            const __bf16* brow = Bc + (size_t)(nw * 128 + t * 16 + ln) * 256;
            acc[t] = wmma_bf16(af, load_b_frag(brow, k0, hi), acc[t]);
        }
    }
#pragma unroll
    for (int t = 0; t < 8; ++t) {
        int gc = nw * 128 + t * 16 + ln;
        float bv = bias[gc];
#pragma unroll
        for (int r = 0; r < 8; ++r) {
            int gm = gm0 + hi * 8 + r;
            h[(size_t)gm * 256 + gc] = acc[t][r] + bv;
        }
    }
}

// ---------- row L2-normalize: nb = h / max(||h||, eps), bf16 out (1 wave / row) ----------
__global__ void k_normalize(const float* __restrict__ h, __bf16* __restrict__ nb, int rows) {
    int wave = threadIdx.x >> 5, lane = threadIdx.x & 31;
    int row = blockIdx.x * 8 + wave;
    if (row >= rows) return;
    const float* hr = h + (size_t)row * 256;
    float s = 0.0f;
    float v[8];
#pragma unroll
    for (int i = 0; i < 8; ++i) { v[i] = hr[lane + i * 32]; s += v[i] * v[i]; }
#pragma unroll
    for (int m = 16; m >= 1; m >>= 1) s += __shfl_xor(s, m, 32);
    float inv = 1.0f / fmaxf(sqrtf(s), 1e-12f);
    __bf16* nr = nb + (size_t)row * 256;
#pragma unroll
    for (int i = 0; i < 8; ++i) nr[lane + i * 32] = (__bf16)(v[i] * inv);
}

// ---------- fused similarity + exp row-sum (+diag) ----------
// rowsum_part[split][i] = sum over this split's column tiles of exp((A_i . B_j)/tau)
// diag[i] = exp((A_i . B_i)/tau), written by the unique owning workgroup.
__global__ __launch_bounds__(256) void k_sim(const __bf16* __restrict__ A,
                                             const __bf16* __restrict__ B,
                                             float* __restrict__ rowsum_part,
                                             float* __restrict__ diag, int N) {
    __shared__ float rp[4][2][16];
    int tid = threadIdx.x;
    int wave = tid >> 5, lane = tid & 31;
    int mw = wave & 3, nw = wave >> 2;
    int hi = lane >> 4, ln = lane & 15;
    int it = blockIdx.x;
    int split = blockIdx.y;
    int gm0 = it * 64 + mw * 16;
    int NB = N >> 8;  // 256-wide column blocks

    const __bf16* arow = A + (size_t)(gm0 + ln) * 256;
    float rowAcc[8] = {0, 0, 0, 0, 0, 0, 0, 0};

    for (int jb = split; jb < NB; jb += NSPLIT) {
        int gn0 = jb * 256 + nw * 128;
        v8f acc[8] = {};
#pragma unroll
        for (int kc = 0; kc < 8; ++kc) {
            int k0 = kc * 32;
            v16bf af = load_a_frag(arow, k0, hi);
#pragma unroll
            for (int t = 0; t < 8; ++t) {
                const __bf16* brow = B + (size_t)(gn0 + t * 16 + ln) * 256;
                acc[t] = wmma_bf16(af, load_b_frag(brow, k0, hi), acc[t]);
            }
        }
#pragma unroll
        for (int t = 0; t < 8; ++t) {
            int gc = gn0 + t * 16 + ln;
#pragma unroll
            for (int r = 0; r < 8; ++r) {
                float e = __expf(acc[t][r] * INV_TAU);
                rowAcc[r] += e;
                int gm = gm0 + hi * 8 + r;
                if (gm == gc) diag[gm] = e;
            }
        }
    }

    // reduce across the 16 lanes of each half (fixed order -> deterministic)
#pragma unroll
    for (int m = 8; m >= 1; m >>= 1) {
#pragma unroll
        for (int r = 0; r < 8; ++r) rowAcc[r] += __shfl_xor(rowAcc[r], m, 32);
    }
    if (ln == 0) {  // lanes 0 (rows 0..7) and 16 (rows 8..15)
#pragma unroll
        for (int r = 0; r < 8; ++r) rp[mw][nw][hi * 8 + r] = rowAcc[r];
    }
    __syncthreads();
    if (tid < 64) {  // combine the two n-waves in fixed order, store
        int m = tid >> 4, r = tid & 15;
        rowsum_part[(size_t)split * N + it * 64 + tid] = rp[m][0][r] + rp[m][1][r];
    }
}

// ---------- loss: per-row combine + deterministic block reduce ----------
__global__ void k_loss_partial(const float* __restrict__ sp /*[4][NSPLIT][N]*/,
                               const float* __restrict__ d11, const float* __restrict__ d12,
                               const float* __restrict__ d21, const float* __restrict__ d22,
                               float* __restrict__ partials, int N) {
    __shared__ float red[256];
    int i = blockIdx.x * 256 + threadIdx.x;
    float v = 0.0f;
    if (i < N) {
        float S[4];
#pragma unroll
        for (int q = 0; q < 4; ++q) {
            float a = 0.0f;
#pragma unroll
            for (int s = 0; s < NSPLIT; ++s) a += sp[((size_t)q * NSPLIT + s) * N + i];
            S[q] = a;
        }
        float l1 = logf(S[0] + S[1] - d11[i]) - logf(d12[i]);
        float l2 = logf(S[3] + S[2] - d22[i]) - logf(d21[i]);
        v = 0.5f * (l1 + l2);
    }
    red[threadIdx.x] = v;
    __syncthreads();
    for (int s = 128; s >= 1; s >>= 1) {
        if (threadIdx.x < s) red[threadIdx.x] += red[threadIdx.x + s];
        __syncthreads();
    }
    if (threadIdx.x == 0) partials[blockIdx.x] = red[0];
}

__global__ void k_loss_final(const float* __restrict__ partials, int nb, float invN,
                             float* __restrict__ out) {
    if (threadIdx.x == 0) {
        float s = 0.0f;
        for (int i = 0; i < nb; ++i) s += partials[i];
        out[0] = s * invN;
    }
}

// ---------- host launcher ----------
extern "C" void kernel_launch(void* const* d_in, const int* in_sizes, int n_in,
                              void* d_out, int out_size, void* d_ws, size_t ws_size,
                              hipStream_t stream) {
    const float* z1 = (const float*)d_in[0];
    const float* z2 = (const float*)d_in[1];
    const float* W1 = (const float*)d_in[2];
    const float* b1 = (const float*)d_in[3];
    const float* W2 = (const float*)d_in[4];
    const float* b2 = (const float*)d_in[5];

    const int H = 256;
    const int N = in_sizes[0] / H;   // 12288
    const int rows2 = 2 * N;
    const size_t RC = (size_t)rows2 * H;  // elements in a [2N x 256] matrix

    char* ws = (char*)d_ws;
    size_t off = 0;
    __bf16* zb  = (__bf16*)(ws + off); off += RC * sizeof(__bf16);
    __bf16* W1c = (__bf16*)(ws + off); off += (size_t)H * H * sizeof(__bf16);
    __bf16* W2c = (__bf16*)(ws + off); off += (size_t)H * H * sizeof(__bf16);
    __bf16* mid = (__bf16*)(ws + off); off += RC * sizeof(__bf16);
    float*  h   = (float*) (ws + off); off += RC * sizeof(float);
    __bf16* nb  = (__bf16*)(ws + off); off += RC * sizeof(__bf16);
    float*  sp  = (float*) (ws + off); off += (size_t)4 * NSPLIT * N * sizeof(float);
    float*  dg  = (float*) (ws + off); off += (size_t)4 * N * sizeof(float);
    int nlb = (N + 255) / 256;
    float* partials = (float*)(ws + off); off += (size_t)nlb * sizeof(float);

    int ne = N * H;
    k_cvt_bf16<<<(ne + 255) / 256, 256, 0, stream>>>(z1, zb, ne);
    k_cvt_bf16<<<(ne + 255) / 256, 256, 0, stream>>>(z2, zb + (size_t)N * H, ne);
    k_transpose_cvt<<<256, 256, 0, stream>>>(W1, W1c);
    k_transpose_cvt<<<256, 256, 0, stream>>>(W2, W2c);

    k_proj1<<<rows2 / 64, 256, 0, stream>>>(zb, W1c, b1, mid);
    k_proj2<<<rows2 / 64, 256, 0, stream>>>(mid, W2c, b2, h);
    k_normalize<<<(rows2 + 7) / 8, 256, 0, stream>>>(h, nb, rows2);

    const __bf16* n1 = nb;
    const __bf16* n2 = nb + (size_t)N * H;
    size_t spStride = (size_t)NSPLIT * N;
    dim3 simGrid(N / 64, NSPLIT);
    k_sim<<<simGrid, 256, 0, stream>>>(n1, n1, sp + 0 * spStride, dg + 0 * N, N);  // refl 1
    k_sim<<<simGrid, 256, 0, stream>>>(n1, n2, sp + 1 * spStride, dg + 1 * N, N);  // between 1->2
    k_sim<<<simGrid, 256, 0, stream>>>(n2, n1, sp + 2 * spStride, dg + 2 * N, N);  // between 2->1
    k_sim<<<simGrid, 256, 0, stream>>>(n2, n2, sp + 3 * spStride, dg + 3 * N, N);  // refl 2

    k_loss_partial<<<nlb, 256, 0, stream>>>(sp, dg, dg + N, dg + 2 * N, dg + 3 * N, partials, N);
    k_loss_final<<<1, 32, 0, stream>>>(partials, nlb, 1.0f / (float)N, (float*)d_out);
}
